// QuantumSuperpositionLayer_21509196218746
// MI455X (gfx1250) — matrix operand order repaired and checked by reading, not compiled, not run on hardware
//
#include <hip/hip_runtime.h>
#include <math.h>

#define NSTATES 8
#define HID     1024
#define BATCH   16384
#define DT_C    0.01f
#define EPS_N   1e-8f
#define EPS_L   1e-12f

typedef __attribute__((ext_vector_type(16))) __bf16 v16bf;
typedef __attribute__((ext_vector_type(8)))  float  v8f;

union Frag {
    uint4 u4[2];
    v16bf v;
};

__device__ __forceinline__ unsigned short f2bf(float x) {
    unsigned int u = __float_as_uint(x);
    unsigned int r = u + 0x7FFFu + ((u >> 16) & 1u);
    return (unsigned short)(r >> 16);
}
__device__ __forceinline__ float bf2f(unsigned short b) {
    return __uint_as_float(((unsigned int)b) << 16);
}

// Generic pointer to a __shared__ object: low 32 bits are the LDS byte offset
// (flat LDS aperture).
__device__ __forceinline__ unsigned lds_off(const void* p) {
    return (unsigned)(uintptr_t)p;
}

// GLOBAL_LOAD_ASYNC_TO_LDS_B128, GVS mode: mem = SADDR + VADDR(u32 bytes),
// LDS dest = VGPR holding LDS byte address. Tracked by ASYNCcnt.
__device__ __forceinline__ void async_ld_b128(const void* sbase, unsigned voff_bytes,
                                              unsigned lds_byte_addr) {
    asm volatile("global_load_async_to_lds_b128 %0, %1, %2"
                 :
                 : "v"(lds_byte_addr), "v"(voff_bytes), "s"(sbase)
                 : "memory");
}
__device__ __forceinline__ void wait_asynccnt0() {
    asm volatile("s_wait_asynccnt 0x0" ::: "memory");
}

__device__ __forceinline__ float blockReduce256(float v, float* red, int t) {
    red[t] = v;
    __syncthreads();
    for (int off = 128; off > 0; off >>= 1) {
        if (t < off) red[t] += red[t + off];
        __syncthreads();
    }
    float r = red[0];
    __syncthreads();
    return r;
}

// ---------------------------------------------------------------------------
// Kernel 1: normalize amps, U = expm(-i dt H) (Taylor, tiny norm), evolve,
// write w=Re(amps) to ws + all small outputs to d_out tail. One block of 256.
// ---------------------------------------------------------------------------
__global__ __launch_bounds__(256) void qsl_setup(
    const float* __restrict__ AR, const float* __restrict__ AI,
    const float* __restrict__ H,
    float* __restrict__ w_out,          // ws: [8][1024]
    float* __restrict__ out_tail)       // d_out + B*D
{
    const int t = threadIdx.x;
    __shared__ float Aim[NSTATES][NSTATES];
    __shared__ float Ure[NSTATES][NSTATES], Uim[NSTATES][NSTATES];
    __shared__ float Tre[NSTATES][NSTATES], Tim[NSTATES][NSTATES];
    __shared__ float red[256];
    __shared__ float rnorm[NSTATES];
    __shared__ float lamS;

    // A = -i*dt*Hs  (Hs = symmetrized H). A_re = 0, A_im = -dt*Hs.
    if (t < 64) {
        int i = t >> 3, j = t & 7;
        float hs = 0.5f * (H[i * 8 + j] + H[j * 8 + i]);
        float aim = -DT_C * hs;
        Aim[i][j] = aim;
        Ure[i][j] = (i == j) ? 1.0f : 0.0f;  // U = I + A + ...
        Uim[i][j] = aim;
        Tre[i][j] = 0.0f;                    // T = A
        Tim[i][j] = aim;
    }
    __syncthreads();
    for (int n = 2; n <= 12; ++n) {
        float tre = 0.f, tim = 0.f;
        if (t < 64) {
            int i = t >> 3, j = t & 7;
            float sr = 0.f, si = 0.f;
            for (int k = 0; k < 8; ++k) {   // (Tre+iTim)*(i*Aim)
                sr -= Tim[i][k] * Aim[k][j];
                si += Tre[i][k] * Aim[k][j];
            }
            float inv = 1.0f / (float)n;
            tre = sr * inv; tim = si * inv;
        }
        __syncthreads();
        if (t < 64) {
            int i = t >> 3, j = t & 7;
            Tre[i][j] = tre; Tim[i][j] = tim;
            Ure[i][j] += tre; Uim[i][j] += tim;
        }
        __syncthreads();
    }

    // Each thread owns 4 columns d = t + 256*c
    float xr[8][4], xi[8][4];
    for (int c = 0; c < 4; ++c) {
        int d = t + 256 * c;
        for (int s = 0; s < 8; ++s) {
            xr[s][c] = AR[s * HID + d];
            xi[s][c] = AI[s * HID + d];
        }
    }
    // per-state norms over d
    for (int s = 0; s < 8; ++s) {
        float acc = 0.f;
        for (int c = 0; c < 4; ++c)
            acc += xr[s][c] * xr[s][c] + xi[s][c] * xi[s][c];
        float tot = blockReduce256(acc, red, t);
        if (t == 0) rnorm[s] = rsqrtf(tot + EPS_N);
        __syncthreads();
    }

    float* ampdist = out_tail + 10;        // after collapse_prob(8)+entropy+coherence
    float* phase   = ampdist + NSTATES * HID;

    float cp[8]; for (int s = 0; s < 8; ++s) cp[s] = 0.f;
    float lam_acc = 0.f, sabs_acc = 0.f;
    for (int c = 0; c < 4; ++c) {
        int d = t + 256 * c;
        float arn[8], ain[8];
        for (int k = 0; k < 8; ++k) {
            arn[k] = xr[k][c] * rnorm[k];
            ain[k] = xi[k][c] * rnorm[k];
        }
        float p[8]; float colsum = 0.f;
        for (int s = 0; s < 8; ++s) {
            float re = 0.f, im = 0.f;
            for (int k = 0; k < 8; ++k) {
                re += Ure[s][k] * arn[k] - Uim[s][k] * ain[k];
                im += Ure[s][k] * ain[k] + Uim[s][k] * arn[k];
            }
            float pp = re * re + im * im;
            p[s] = pp; colsum += pp;
            w_out[s * HID + d]   = re;
            ampdist[s * HID + d] = pp;
            phase[s * HID + d]   = atan2f(im, re);
            lam_acc  += pp;
            sabs_acc += sqrtf(pp);
        }
        float rinv = 1.0f / (colsum + EPS_N);
        for (int s = 0; s < 8; ++s) cp[s] += p[s] * rinv;
    }

    for (int s = 0; s < 8; ++s) {
        float tot = blockReduce256(cp[s], red, t);
        if (t == 0) out_tail[s] = tot * (1.0f / (float)HID);
        __syncthreads();
    }
    {
        float lam = blockReduce256(lam_acc, red, t);
        if (t == 0) lamS = lam;
        __syncthreads();
        float sab = blockReduce256(sabs_acc, red, t);
        if (t == 0) {
            float l = lamS;
            out_tail[8] = -(l * logf(l + EPS_L));   // entanglement entropy
            out_tail[9] = sab * sab - l;            // coherence
        }
    }
}

// ---------------------------------------------------------------------------
// Kernel 2: M[d,e] = sum_s w[s,e] * P[s,d,e]; write transposed Mt[e][d] as
// bf16 hi/lo split (for the bf16x3 WMMA GEMM). 64x64 tiles, LDS transpose.
// ---------------------------------------------------------------------------
__global__ __launch_bounds__(256) void qsl_build_m(
    const float* __restrict__ P, const float* __restrict__ w,
    unsigned short* __restrict__ MtHi, unsigned short* __restrict__ MtLo)
{
    const int t  = threadIdx.x;
    const int e0 = blockIdx.x * 64;
    const int d0 = blockIdx.y * 64;
    __shared__ float          wt[NSTATES][64];
    __shared__ unsigned short tHi[64][72];
    __shared__ unsigned short tLo[64][72];

    for (int idx = t; idx < NSTATES * 64; idx += 256) {
        int s = idx >> 6, e = idx & 63;
        wt[s][e] = w[s * HID + e0 + e];
    }
    __syncthreads();

    const int tx = t & 63, ty = t >> 6;       // tx: e-lane (coalesced reads)
    for (int i = 0; i < 16; ++i) {
        int dl = ty + 4 * i;
        int d  = d0 + dl;
        float m = 0.f;
        for (int s = 0; s < 8; ++s)
            m += wt[s][tx] * P[((size_t)s * HID + d) * HID + e0 + tx];
        unsigned short hi = f2bf(m);
        unsigned short lo = f2bf(m - bf2f(hi));
        tHi[dl][tx] = hi;
        tLo[dl][tx] = lo;
    }
    __syncthreads();
    for (int i = 0; i < 16; ++i) {            // write transposed, coalesced in d
        int el = ty + 4 * i;
        MtHi[(size_t)(e0 + el) * HID + d0 + tx] = tHi[tx][el];
        MtLo[(size_t)(e0 + el) * HID + d0 + tx] = tLo[tx][el];
    }
}

// ---------------------------------------------------------------------------
// Kernel 3: C[16384,1024] = X @ M via V_WMMA_F32_16X16X32_BF16, bf16x3 split.
// Block tile 128x64, 8 waves of 32x32 (2x2 16x16 tiles), K-step 32.
// Double-buffered LDS; B tiles staged with GLOBAL_LOAD_ASYNC_TO_LDS_B128
// (ASYNCcnt), A tiles converted f32->bf16 hi/lo in VALU under the WMMAs.
// ---------------------------------------------------------------------------
__global__ __launch_bounds__(256) void qsl_gemm(
    const float* __restrict__ X,               // [16384][1024] f32
    const unsigned short* __restrict__ BtHi,   // Mt: [1024 e][1024 d] bf16
    const unsigned short* __restrict__ BtLo,
    float* __restrict__ C)                     // [16384][1024]
{
    const int t      = threadIdx.x;
    const int rowBlk = blockIdx.y * 128;
    const int colBlk = blockIdx.x * 64;

    __shared__ unsigned short sAhi[2][128][40];  // stride 80B: 16B-aligned
    __shared__ unsigned short sAlo[2][128][40];
    __shared__ unsigned short sBhi[2][64][40];
    __shared__ unsigned short sBlo[2][64][40];

    const int la_r = t >> 3;                   // 0..31
    const int la_c = (t & 7) * 4;              // float4 per thread
    const int lb_r = t >> 2;                   // 0..63
    const int lb_c = (t & 3) * 8;              // 8 bf16 = 16B chunk

    const int wave  = t >> 5;
    const int lane  = t & 31;
    const int wr    = (wave & 3) * 32;
    const int wc    = (wave >> 2) * 32;
    const int lhalf = lane >> 4;               // 0/1
    const int lmod  = lane & 15;

    const size_t   aBase  = (size_t)(rowBlk + la_r) * HID + la_c;
    const unsigned bBase  = ((unsigned)(colBlk + lb_r) * HID + (unsigned)lb_c) * 2u; // bytes

    v8f acc[2][2];
    for (int i = 0; i < 2; ++i)
        for (int j = 0; j < 2; ++j)
            acc[i][j] = (v8f)(0.0f);

    // ---- A-tile stage: convert 4 float4 rows to bf16 hi/lo and ds-write ----
    auto stageA = [&](int buf, const float4* xv) {
        for (int i = 0; i < 4; ++i) {
            int r = la_r + 32 * i;
            float xs[4] = {xv[i].x, xv[i].y, xv[i].z, xv[i].w};
            unsigned short h[4], l[4];
            for (int c = 0; c < 4; ++c) {
                h[c] = f2bf(xs[c]);
                l[c] = f2bf(xs[c] - bf2f(h[c]));
            }
            uint2 uh, ul;
            uh.x = (unsigned)h[0] | ((unsigned)h[1] << 16);
            uh.y = (unsigned)h[2] | ((unsigned)h[3] << 16);
            ul.x = (unsigned)l[0] | ((unsigned)l[1] << 16);
            ul.y = (unsigned)l[2] | ((unsigned)l[3] << 16);
            *(uint2*)&sAhi[buf][r][la_c] = uh;
            *(uint2*)&sAlo[buf][r][la_c] = ul;
        }
    };

    // ---- prologue: stage 0 into buffer 0 ----
    {
        float4 xv[4];
        for (int i = 0; i < 4; ++i)
            xv[i] = *(const float4*)&X[aBase + (size_t)(32 * i) * HID];
        async_ld_b128(BtHi, bBase, lds_off(&sBhi[0][lb_r][lb_c]));
        async_ld_b128(BtLo, bBase, lds_off(&sBlo[0][lb_r][lb_c]));
        stageA(0, xv);
        wait_asynccnt0();
        __syncthreads();
    }

    for (int kt = 0; kt < HID / 32; ++kt) {
        const int  cur     = kt & 1;
        const int  nxt     = cur ^ 1;
        const bool hasNext = (kt + 1) < HID / 32;

        // issue next-stage global traffic first (hides under the WMMAs)
        float4 xv[4];
        if (hasNext) {
            const int k0n = (kt + 1) * 32;
            for (int i = 0; i < 4; ++i)
                xv[i] = *(const float4*)&X[aBase + (size_t)(32 * i) * HID + k0n];
            async_ld_b128(BtHi, bBase + (unsigned)k0n * 2u,
                          lds_off(&sBhi[nxt][lb_r][lb_c]));
            async_ld_b128(BtLo, bBase + (unsigned)k0n * 2u,
                          lds_off(&sBlo[nxt][lb_r][lb_c]));
        }

        // gather fragments per ISA layout from current buffer
        Frag aH[2], aL[2], bH[2], bL[2];
        for (int i = 0; i < 2; ++i) {
            int r  = wr + i * 16 + lmod;
            int kk = 8 * lhalf;                        // K = 8*lhalf+0..7, then +16
            aH[i].u4[0] = *(const uint4*)&sAhi[cur][r][kk];
            aH[i].u4[1] = *(const uint4*)&sAhi[cur][r][kk + 16];
            aL[i].u4[0] = *(const uint4*)&sAlo[cur][r][kk];
            aL[i].u4[1] = *(const uint4*)&sAlo[cur][r][kk + 16];
        }
        for (int j = 0; j < 2; ++j) {
            int e  = wc + j * 16 + lmod;
            int kk = 8 * lhalf;                        // K = 8*lhalf+0..7, then +16
            bH[j].u4[0] = *(const uint4*)&sBhi[cur][e][kk];
            bH[j].u4[1] = *(const uint4*)&sBhi[cur][e][kk + 16];
            bL[j].u4[0] = *(const uint4*)&sBlo[cur][e][kk];
            bL[j].u4[1] = *(const uint4*)&sBlo[cur][e][kk + 16];
        }

        for (int i = 0; i < 2; ++i)
            for (int j = 0; j < 2; ++j) {
                acc[i][j] = __builtin_amdgcn_wmma_f32_16x16x32_bf16(
                    false, aH[i].v, false, bH[j].v, (short)0, acc[i][j], false, false);
                acc[i][j] = __builtin_amdgcn_wmma_f32_16x16x32_bf16(
                    false, aH[i].v, false, bL[j].v, (short)0, acc[i][j], false, false);
                acc[i][j] = __builtin_amdgcn_wmma_f32_16x16x32_bf16(
                    false, aL[i].v, false, bH[j].v, (short)0, acc[i][j], false, false);
            }

        if (hasNext) {
            stageA(nxt, xv);      // VALU convert + ds_write under WMMA shadow
            wait_asynccnt0();     // B tiles for stage kt+1 landed in LDS
            __syncthreads();
        }
    }

    // C layout: VGPR r -> row r + 8*lhalf, col lmod
    for (int i = 0; i < 2; ++i)
        for (int j = 0; j < 2; ++j) {
            int col   = colBlk + wc + j * 16 + lmod;
            int rbase = rowBlk + wr + i * 16 + 8 * lhalf;
            for (int r = 0; r < 8; ++r)
                C[(size_t)(rbase + r) * HID + col] = acc[i][j][r];
        }
}

// ---------------------------------------------------------------------------
extern "C" void kernel_launch(void* const* d_in, const int* in_sizes, int n_in,
                              void* d_out, int out_size, void* d_ws, size_t ws_size,
                              hipStream_t stream) {
    (void)in_sizes; (void)n_in; (void)out_size; (void)ws_size;
    const float* X  = (const float*)d_in[0];   // input_state [16384,1024]
    const float* AR = (const float*)d_in[1];   // amplitude_real [8,1024]
    const float* AI = (const float*)d_in[2];   // amplitude_imag [8,1024]
    const float* H  = (const float*)d_in[3];   // hamiltonian [8,8]
    const float* P  = (const float*)d_in[4];   // projections [8,1024,1024]

    float* out      = (float*)d_out;
    float* out_tail = out + (size_t)BATCH * HID;   // collapse_prob, S, C, dist, phase

    float*          w_ws = (float*)d_ws;                                   // 32 KB
    unsigned short* MtHi = (unsigned short*)((char*)d_ws + 32768);         // 2 MB
    unsigned short* MtLo = (unsigned short*)((char*)d_ws + 32768 + 2u * 1024 * 1024);

    qsl_setup<<<1, 256, 0, stream>>>(AR, AI, H, w_ws, out_tail);

    dim3 g2(HID / 64, HID / 64);
    qsl_build_m<<<g2, 256, 0, stream>>>(P, w_ws, MtHi, MtLo);

    dim3 g3(HID / 64, BATCH / 128);
    qsl_gemm<<<g3, 256, 0, stream>>>(X, MtHi, MtLo, out);
}
